// MultilayerGRU_28535762715266
// MI455X (gfx1250) — matrix operand order repaired and checked by reading, not compiled, not run on hardware
//
#include <hip/hip_runtime.h>
#include <hip/hip_bf16.h>
#include <stdint.h>

typedef __attribute__((ext_vector_type(8)))  float  v8f;
typedef __attribute__((ext_vector_type(8)))  __bf16 v8y;
typedef __attribute__((ext_vector_type(16))) __bf16 v16y;

#define GRU_B 64
#define GRU_S 512
#define GRU_I 256
#define GRU_H 1024
#define GRU_O 256
#define GRU_G 3072   // 3*H
#define NBLK  64     // persistent grid: 64 blocks x 128 threads = 256 waves

#if defined(__has_builtin)
#  if __has_builtin(__builtin_amdgcn_global_load_async_to_lds_b128) && \
      __has_builtin(__builtin_amdgcn_s_wait_asynccnt)
#    define HAVE_ASYNC 1
#  endif
#endif
#ifndef HAVE_ASYNC
#  define HAVE_ASYNC 0
#endif

#if HAVE_ASYNC
typedef int v4i_ __attribute__((vector_size(16)));
typedef __attribute__((address_space(1))) v4i_* gv4i_p;   // global int32x4*
typedef __attribute__((address_space(3))) v4i_* lv4i_p;   // LDS int32x4*
#endif

__device__ __forceinline__ __bf16 f2bf(float f) {
  uint32_t u = __builtin_bit_cast(uint32_t, f);
  uint32_t r = u + 0x7FFFu + ((u >> 16) & 1u);   // round-to-nearest-even
  uint16_t s = (uint16_t)(r >> 16);
  return __builtin_bit_cast(__bf16, s);
}

__device__ __forceinline__ float sigmoidf_(float x) { return 1.0f / (1.0f + __expf(-x)); }
__device__ __forceinline__ float tanhf_(float x)    { return 2.0f * sigmoidf_(2.0f * x) - 1.0f; }

// ---------------------------------------------------------------------------
// Pack a KxN f32 weight into WMMA bf16 B-operand order:
// tile (kt,nt): 32 lanes x 16 halfs; lane's values =
// W[kt*32 + (lane>=16?16:0) + j][nt*16 + (lane&15)], j=0..15. Tile idx = nt*Kt+kt.
// ---------------------------------------------------------------------------
__global__ void pack_weight(const float* __restrict__ src, __bf16* __restrict__ dst,
                            int K, int N) {
  int gid = blockIdx.x * blockDim.x + threadIdx.x;
  int Kt = K >> 5;
  int total = Kt * (N >> 4) * 32;
  if (gid >= total) return;
  int lane = gid & 31;
  int tile = gid >> 5;
  int kt = tile % Kt;
  int nt = tile / Kt;
  int col = nt * 16 + (lane & 15);
  int kb  = kt * 32 + ((lane >> 4) << 4);
  __bf16* o = dst + (size_t)gid * 16;
#pragma unroll
  for (int j = 0; j < 16; ++j)
    o[j] = f2bf(src[(size_t)(kb + j) * N + col]);
}

__global__ void sync_init(int* s) {
  if (threadIdx.x < 2) s[threadIdx.x] = 0;
}

// device-wide barrier: counter + generation, agent-scope atomics
__device__ __forceinline__ void gbar(int* cnt, int* gen) {
  __syncthreads();
  if (threadIdx.x == 0) {
    __threadfence();
    int g = __hip_atomic_load(gen, __ATOMIC_RELAXED, __HIP_MEMORY_SCOPE_AGENT);
    int v = __hip_atomic_fetch_add(cnt, 1, __ATOMIC_ACQ_REL, __HIP_MEMORY_SCOPE_AGENT);
    if (v == NBLK - 1) {
      __hip_atomic_store(cnt, 0, __ATOMIC_RELAXED, __HIP_MEMORY_SCOPE_AGENT);
      __hip_atomic_fetch_add(gen, 1, __ATOMIC_RELEASE, __HIP_MEMORY_SCOPE_AGENT);
    } else {
      while (__hip_atomic_load(gen, __ATOMIC_ACQUIRE, __HIP_MEMORY_SCOPE_AGENT) == g)
        __builtin_amdgcn_s_sleep(1);
    }
  }
  __syncthreads();
}

// A-fragment, contiguous k order: lane half h holds k = 16h + s
__device__ __forceinline__ v16y load_a(const __bf16* __restrict__ A, int row, int ld,
                                       int kb, int half) {
  const __bf16* p = A + (size_t)row * ld + kb + half * 16;
  v8y lo = *(const v8y*)p;
  v8y hi = *(const v8y*)(p + 8);
  v16y a;
#pragma unroll
  for (int i = 0; i < 8; ++i) { a[i] = lo[i]; a[8 + i] = hi[i]; }
  return a;
}

#define WMMA_BF16(a, b, c) \
  __builtin_amdgcn_wmma_f32_16x16x32_bf16(false, (a), false, (b), (short)0, (c), false, false)

#if HAVE_ASYNC
// wave 0 stages 3 B tiles (z,r,g; 3 KB) for k-chunk kt into LDS buffer `buf`
// via 6 async-to-LDS b128 ops (each moves 32 lanes x 16B = 512B). ASYNCcnt-tracked.
__device__ __forceinline__ void stage_b(const __bf16* __restrict__ Wp, int Kt, int kt,
                                        int buf, int lane, int nt, __bf16* sB) {
  uint32_t ldsbase = (uint32_t)(uintptr_t)sB + (uint32_t)buf * 3072u;
#pragma unroll
  for (int i = 0; i < 6; ++i) {
    int c = i * 32 + lane;           // chunk 0..191, 16B each
    int gate = c >> 6, cc = c & 63;
    const __bf16* src = Wp + ((size_t)(nt + (gate << 6)) * Kt + kt) * 512 + cc * 8;
    __builtin_amdgcn_global_load_async_to_lds_b128(
        (gv4i_p)(uintptr_t)src,
        (lv4i_p)(uintptr_t)(ldsbase + (uint32_t)c * 16u),
        0, 0);
  }
}
#endif

// one M-tile x 3-gate GEMM chain: acc{z,r,g} += A(16xK) * W(Kx16 per gate)
__device__ __forceinline__ void gemm3(const __bf16* __restrict__ A, int K,
                                      const __bf16* __restrict__ Wp,
                                      v8f& az, v8f& ar, v8f& ag,
                                      __bf16* sB, int lane, int wid, int nt,
                                      int row0, int half) {
  const int Kt = K >> 5;
#if HAVE_ASYNC
  if (wid == 0) stage_b(Wp, Kt, 0, 0, lane, nt, sB);
  for (int kt = 0; kt < Kt; ++kt) {
    const int cb = kt & 1;
    if (wid == 0) {
      if (kt + 1 < Kt) {
        stage_b(Wp, Kt, kt + 1, cb ^ 1, lane, nt, sB);
        __builtin_amdgcn_s_wait_asynccnt(6);   // kt's 6 loads complete (in-order)
      } else {
        __builtin_amdgcn_s_wait_asynccnt(0);
      }
    }
    __syncthreads();                           // LDS buffer cb valid for all waves
    v16y a  = load_a(A, row0, K, kt * 32, half);
    const __bf16* base = sB + cb * 1536;
    v16y wz = *(const v16y*)(base +         lane * 16);
    v16y wr = *(const v16y*)(base +  512 +  lane * 16);
    v16y wg = *(const v16y*)(base + 1024 +  lane * 16);
    az = WMMA_BF16(a, wz, az);
    ar = WMMA_BF16(a, wr, ar);
    ag = WMMA_BF16(a, wg, ag);
    __syncthreads();                           // done reading cb before re-stage
  }
#else
  const __bf16* bz = Wp + (size_t)((nt      ) * Kt) * 512 + lane * 16;
  const __bf16* br = Wp + (size_t)((nt +  64) * Kt) * 512 + lane * 16;
  const __bf16* bg = Wp + (size_t)((nt + 128) * Kt) * 512 + lane * 16;
  for (int kt = 0; kt < Kt; ++kt) {
    __builtin_prefetch(bz + (size_t)(kt + 2) * 512, 0, 0);
    __builtin_prefetch(br + (size_t)(kt + 2) * 512, 0, 0);
    __builtin_prefetch(bg + (size_t)(kt + 2) * 512, 0, 0);
    v16y a  = load_a(A, row0, K, kt * 32, half);
    v16y wz = *(const v16y*)(bz + (size_t)kt * 512);
    v16y wr = *(const v16y*)(br + (size_t)kt * 512);
    v16y wg = *(const v16y*)(bg + (size_t)kt * 512);
    az = WMMA_BF16(a, wz, az);
    ar = WMMA_BF16(a, wr, ar);
    ag = WMMA_BF16(a, wg, ag);
  }
#endif
}

// ---------------------------------------------------------------------------
// Persistent cooperative kernel: whole GRU in one launch.
// Block = one N-tile (nt=blockIdx.x, 0..63); its 4 waves = the 4 M-tiles and
// share B fragments through LDS. Device barrier between layer phases.
// ---------------------------------------------------------------------------
__global__ void __launch_bounds__(128) gru_persistent(
    const float* __restrict__ x, const float* __restrict__ h0,
    const __bf16* __restrict__ Wx0p, const __bf16* __restrict__ Wh0p,
    const float* __restrict__ bh0,
    const __bf16* __restrict__ Wx1p, const __bf16* __restrict__ Wh1p,
    const __bf16* __restrict__ Wx2p, const __bf16* __restrict__ Wh2p,
    const float* __restrict__ bh,
    const __bf16* __restrict__ Wyp, const float* __restrict__ by,
    float* __restrict__ hf, __bf16* __restrict__ hb, __bf16* __restrict__ xt,
    float* __restrict__ out, int* __restrict__ sync) {
  const int tid  = threadIdx.x;
  const int lane = tid & 31;
  const int wid  = tid >> 5;        // wave in block = M-tile in layer phases
  const int nt   = blockIdx.x;      // N-tile
  int* cnt = sync;
  int* gen = sync + 1;

  __shared__ __align__(16) __bf16 sB[2 * 1536];   // double-buffered 3-tile B stage
  __shared__ float sRed[4 * 256];                 // projection K-split reduce

  const int half = lane >> 4;
  const int row0 = wid * 16 + (lane & 15);
  const int col  = nt * 16 + (lane & 15);

  // ---- init: h0 -> state buffers (buf 0), x[:,0,:] -> xt ----
  for (int gid = nt * 128 + tid; gid < GRU_B * 3 * GRU_H; gid += NBLK * 128) {
    int j = gid % GRU_H, l = (gid / GRU_H) % 3, b = gid / (GRU_H * 3);
    float v = h0[gid];
    size_t off = ((size_t)l * 2) * GRU_B * GRU_H + (size_t)b * GRU_H + j;
    hf[off] = v;
    hb[off] = f2bf(v);
  }
  for (int gid = nt * 128 + tid; gid < GRU_B * GRU_I; gid += NBLK * 128)
    xt[gid] = f2bf(x[(size_t)(gid / GRU_I) * GRU_S * GRU_I + (gid % GRU_I)]);
  gbar(cnt, gen);

  for (int t = 0; t < GRU_S; ++t) {
    const int rb = t & 1, wb = rb ^ 1;
#pragma unroll 1
    for (int l = 0; l < 3; ++l) {
      const __bf16* cur; int Kx; const __bf16* Wxp; const __bf16* Whp; const float* bias;
      if (l == 0) { cur = xt;                                    Kx = GRU_I; Wxp = Wx0p; Whp = Wh0p; bias = bh0; }
      else if (l == 1) { cur = hb + ((size_t)0 * 2 + wb) * GRU_B * GRU_H; Kx = GRU_H; Wxp = Wx1p; Whp = Wh1p; bias = bh; }
      else { cur = hb + ((size_t)1 * 2 + wb) * GRU_B * GRU_H;    Kx = GRU_H; Wxp = Wx2p; Whp = Wh2p; bias = bh + GRU_G; }
      const float*  hpf = hf + ((size_t)l * 2 + rb) * GRU_B * GRU_H;
      const __bf16* hpb = hb + ((size_t)l * 2 + rb) * GRU_B * GRU_H;
      float*  hnf = hf + ((size_t)l * 2 + wb) * GRU_B * GRU_H;
      __bf16* hnb = hb + ((size_t)l * 2 + wb) * GRU_B * GRU_H;

      v8f axz = {}, axr = {}, axg = {}, ahz = {}, ahr = {}, ahg = {};
      gemm3(cur, Kx,    Wxp, axz, axr, axg, sB, lane, wid, nt, row0, half);
      gemm3(hpb, GRU_H, Whp, ahz, ahr, ahg, sB, lane, wid, nt, row0, half);

      const float bz = bias[col], br = bias[GRU_H + col], bg = bias[2 * GRU_H + col];
#pragma unroll
      for (int v = 0; v < 8; ++v) {
        int rowm = wid * 16 + v + 8 * half;
        float z = sigmoidf_(axz[v] + ahz[v] + bz);
        float r = sigmoidf_(axr[v] + ahr[v] + br);
        float g = tanhf_(axg[v] + r * ahg[v] + bg);
        float hp = hpf[(size_t)rowm * GRU_H + col];
        float hn = z * hp + (1.0f - z) * g;
        hnf[(size_t)rowm * GRU_H + col] = hn;
        hnb[(size_t)rowm * GRU_H + col] = f2bf(hn);
      }
      gbar(cnt, gen);
    }

    // ---- phase: y_t projection (one tile per block, K split over 4 waves)
    //      + convert x[:,t+1,:] for the next step ----
    if (t + 1 < GRU_S)
      for (int gid = nt * 128 + tid; gid < GRU_B * GRU_I; gid += NBLK * 128)
        xt[gid] = f2bf(x[((size_t)(gid / GRU_I) * GRU_S + (t + 1)) * GRU_I + (gid % GRU_I)]);
    {
      const __bf16* h2 = hb + ((size_t)2 * 2 + wb) * GRU_B * GRU_H;
      const int mt = nt & 3, pnt = nt >> 2;
      const int prow0 = mt * 16 + (lane & 15);
      v8f acc = {};
      const __bf16* bp = Wyp + (size_t)(pnt * 32) * 512 + lane * 16;
      for (int kt = wid * 8; kt < wid * 8 + 8; ++kt) {
        v16y a = load_a(h2, prow0, GRU_H, kt * 32, half);
        v16y b = *(const v16y*)(bp + (size_t)kt * 512);
        acc = WMMA_BF16(a, b, acc);
      }
#pragma unroll
      for (int v = 0; v < 8; ++v) sRed[wid * 256 + lane * 8 + v] = acc[v];
      __syncthreads();
      if (wid == 0) {
        int pcol = pnt * 16 + (lane & 15);
        float bb = by[pcol];
#pragma unroll
        for (int v = 0; v < 8; ++v) {
          float s = sRed[lane * 8 + v] + sRed[256 + lane * 8 + v] +
                    sRed[512 + lane * 8 + v] + sRed[768 + lane * 8 + v];
          int b_ = mt * 16 + v + 8 * half;
          out[((size_t)b_ * GRU_S + t) * GRU_O + pcol] = s + bb;
        }
      }
    }
    gbar(cnt, gen);
  }

  // ---- final hidden state (B,L,H); newest state is in buffer 0 after t=511 ----
  float* outHid = out + (size_t)GRU_B * GRU_S * GRU_O;
  for (int gid = nt * 128 + tid; gid < GRU_B * 3 * GRU_H; gid += NBLK * 128) {
    int j = gid % GRU_H, l = (gid / GRU_H) % 3, b = gid / (GRU_H * 3);
    outHid[gid] = hf[((size_t)l * 2 + 0) * GRU_B * GRU_H + (size_t)b * GRU_H + j];
  }
}

extern "C" void kernel_launch(void* const* d_in, const int* in_sizes, int n_in,
                              void* d_out, int out_size, void* d_ws, size_t ws_size,
                              hipStream_t stream) {
  const float* x   = (const float*)d_in[0];
  const float* h0  = (const float*)d_in[1];
  const float* Wx0 = (const float*)d_in[2];
  const float* Wh0 = (const float*)d_in[3];
  const float* bh0 = (const float*)d_in[4];
  const float* Wx  = (const float*)d_in[5];
  const float* Wh  = (const float*)d_in[6];
  const float* bh  = (const float*)d_in[7];
  const float* Wy  = (const float*)d_in[8];
  const float* by  = (const float*)d_in[9];
  float* out = (float*)d_out;

  char* ws = (char*)d_ws;
  size_t cur_off = 0;
  auto alloc = [&](size_t bytes) -> char* {
    char* p = ws + cur_off;
    cur_off += (bytes + 255) & ~(size_t)255;
    return p;
  };
  __bf16* Wx0p = (__bf16*)alloc((size_t)GRU_I * GRU_G * 2);
  __bf16* Wh0p = (__bf16*)alloc((size_t)GRU_H * GRU_G * 2);
  __bf16* Wx1p = (__bf16*)alloc((size_t)GRU_H * GRU_G * 2);
  __bf16* Wh1p = (__bf16*)alloc((size_t)GRU_H * GRU_G * 2);
  __bf16* Wx2p = (__bf16*)alloc((size_t)GRU_H * GRU_G * 2);
  __bf16* Wh2p = (__bf16*)alloc((size_t)GRU_H * GRU_G * 2);
  __bf16* Wyp  = (__bf16*)alloc((size_t)GRU_H * GRU_O * 2);
  float*  hf   = (float*) alloc((size_t)3 * 2 * GRU_B * GRU_H * 4);
  __bf16* hb   = (__bf16*)alloc((size_t)3 * 2 * GRU_B * GRU_H * 2);
  __bf16* xt   = (__bf16*)alloc((size_t)GRU_B * GRU_I * 2);
  int*    sync = (int*)   alloc(256);

  pack_weight<<<( 8 * 192 * 32 + 255) / 256, 256, 0, stream>>>(Wx0, Wx0p, GRU_I, GRU_G);
  pack_weight<<<(32 * 192 * 32 + 255) / 256, 256, 0, stream>>>(Wh0, Wh0p, GRU_H, GRU_G);
  pack_weight<<<(32 * 192 * 32 + 255) / 256, 256, 0, stream>>>(Wx, Wx1p, GRU_H, GRU_G);
  pack_weight<<<(32 * 192 * 32 + 255) / 256, 256, 0, stream>>>(Wx + (size_t)GRU_H * GRU_G, Wx2p, GRU_H, GRU_G);
  pack_weight<<<(32 * 192 * 32 + 255) / 256, 256, 0, stream>>>(Wh, Wh1p, GRU_H, GRU_G);
  pack_weight<<<(32 * 192 * 32 + 255) / 256, 256, 0, stream>>>(Wh + (size_t)GRU_H * GRU_G, Wh2p, GRU_H, GRU_G);
  pack_weight<<<(32 * 16 * 32 + 255) / 256, 256, 0, stream>>>(Wy, Wyp, GRU_H, GRU_O);
  sync_init<<<1, 64, 0, stream>>>(sync);

  gru_persistent<<<NBLK, 128, 0, stream>>>(
      x, h0, Wx0p, Wh0p, bh0, Wx1p, Wh1p, Wx2p, Wh2p, bh, Wyp, by,
      hf, hb, xt, out, sync);
}
